// RT_Layer_87668872446685
// MI455X (gfx1250) — compile-verified, matrix operand order checked by reading, not run on hardware
//
#include <hip/hip_runtime.h>
#include <math.h>

// ---------------------------------------------------------------------------
// Deep Hough Transform block for MI455X (gfx1250, wave32, WMMA).
//   conv1x1(BN,ReLU) -> DHT segment-sum -> conv3x3(BN,ReLU) -> conv3x3(BN,ReLU)
// Channels-last bf16 internally. GEMMs: v_wmma_f32_16x16x32_bf16.
// Block = 8 waves = all 8 cout tiles of one spatial group; the shared B slab
// is staged once in LDS (padded stride -> conflict-free ds_load_b128), halo
// zero-filled so the WMMA main loop is branch-free with EXEC all-ones.
// DHT: LDS f32 histograms via ds_add_f32, exclusive stores, no global atomics.
// ---------------------------------------------------------------------------

typedef __attribute__((ext_vector_type(16))) __bf16 v16bf;
typedef __attribute__((ext_vector_type(8)))  __bf16 v8bf;
typedef __attribute__((ext_vector_type(8)))  float  v8f;

#define kN   8
#define kCi  256
#define kD   128
#define kH   128
#define kW   128
#define kHW  (kH * kW)        // 16384
#define kA   180
#define kR   180
#define kAR  (kA * kR)        // 32400
#define kEPS 1e-5f

// Fragment: 16 bf16 for one lane of a 16x32 WMMA operand (ISA 7.12.2 layout):
// elements 0..7 = K base..+7 (16B), elements 8..15 = +16 (next 16B).
__device__ __forceinline__ v16bf frag16(const __bf16* p) {
  v8bf lo = *(const v8bf*)p;
  v8bf hi = *(const v8bf*)(p + 16);
  return __builtin_shufflevector(lo, hi, 0, 1, 2, 3, 4, 5, 6, 7,
                                         8, 9, 10, 11, 12, 13, 14, 15);
}

// ---- BN folding: y = x*scale + shift --------------------------------------
__global__ void bn_precompute(const float* __restrict__ b, const float* __restrict__ g,
                              const float* __restrict__ be, const float* __restrict__ m,
                              const float* __restrict__ v,
                              float* __restrict__ sc, float* __restrict__ sh) {
  int i = blockIdx.x * blockDim.x + threadIdx.x;
  if (i < kD) {
    float s = g[i] * rsqrtf(v[i] + kEPS);
    sc[i] = s;
    sh[i] = (b[i] - m[i]) * s + be[i];
  }
}

// ---- weight packing --------------------------------------------------------
__global__ void pack_f32_to_bf16(const float* __restrict__ src,
                                 __bf16* __restrict__ dst, int n) {
  int i = blockIdx.x * blockDim.x + threadIdx.x;
  if (i < n) dst[i] = (__bf16)src[i];
}

// src [co][ci][3][3] f32 -> dst [tap][co][ci] bf16  (tap = ky*3+kx)
__global__ void pack_w3x3(const float* __restrict__ src, __bf16* __restrict__ dst) {
  int i = blockIdx.x * blockDim.x + threadIdx.x;
  if (i >= kD * kD * 9) return;
  int t  = i % 9;
  int ci = (i / 9) % kD;
  int co = i / (9 * kD);
  dst[((size_t)t * kD + co) * kD + ci] = (__bf16)src[i];
}

// ---- x: NCHW f32 -> channels-last bf16 xt[n][hw][ci], 32x32 LDS tiles ------
__global__ void transpose_x(const float* __restrict__ x, __bf16* __restrict__ xt) {
  __shared__ float tile[32][33];
  const int b   = blockIdx.x;
  const int hwT = b % (kHW / 32);                 // 512
  const int ciT = (b / (kHW / 32)) % (kCi / 32);  // 8
  const int n   = b / ((kHW / 32) * (kCi / 32));
  const int tx = threadIdx.x & 31, ty = threadIdx.x >> 5;
  const float* src = x + ((size_t)n * kCi + ciT * 32) * kHW + hwT * 32;
#pragma unroll
  for (int r = ty; r < 32; r += 8) tile[r][tx] = src[(size_t)r * kHW + tx];
  __syncthreads();
  __bf16* dst = xt + ((size_t)n * kHW + hwT * 32) * kCi + ciT * 32;
#pragma unroll
  for (int r = ty; r < 32; r += 8) dst[(size_t)r * kCi + tx] = (__bf16)tile[tx][r];
}

// ---- conv1: 1x1 conv GEMM, block = (n, 64-pixel group) x 8 cout-tile waves -
// B slab (64 px x 256 ci, padded stride 264) staged once; A from global (L2).
__global__ void __launch_bounds__(256)
conv1_wmma(const __bf16* __restrict__ xt, const __bf16* __restrict__ w1b,
           const float* __restrict__ sc, const float* __restrict__ sh,
           __bf16* __restrict__ fout) {
  __shared__ __bf16 slab[64][264];                 // 33.8 KB, stride 132 dwords
  const int tid = threadIdx.x;
  const int pg = blockIdx.x & 255;                 // 256 pixel groups of 64
  const int n  = blockIdx.x >> 8;

  const __bf16* src = xt + ((size_t)n * kHW + pg * 64) * kCi;
  for (int idx = tid; idx < 64 * 32; idx += 256) { // v8bf chunks
    const int p = idx >> 5, k8 = (idx & 31) * 8;
    *(v8bf*)&slab[p][k8] = *(const v8bf*)(src + (size_t)p * kCi + k8);
  }
  __syncthreads();

  const int wave = tid >> 5, lane = tid & 31;
  const int m0 = wave << 4;
  const int hi = lane >> 4, colj = lane & 15;
  const __bf16* wrow = w1b + (size_t)(m0 + colj) * kCi + hi * 8;

  v8f c[4] = {{}, {}, {}, {}};
#pragma unroll
  for (int kc = 0; kc < kCi / 32; ++kc) {
    const int k0 = kc * 32;
    v16bf a = frag16(wrow + k0);
#pragma unroll
    for (int j = 0; j < 4; ++j) {
      v16bf b = frag16(&slab[colj + j * 16][k0 + hi * 8]);   // ds_load_b128 x2
      c[j] = __builtin_amdgcn_wmma_f32_16x16x32_bf16(false, a, false, b,
                                                     (short)0, c[j], false, false);
    }
  }
#pragma unroll
  for (int j = 0; j < 4; ++j) {
    const int pix = pg * 64 + j * 16 + colj;
    v8bf o;
#pragma unroll
    for (int v = 0; v < 8; ++v) {
      const int m = m0 + v + 8 * hi;
      o[v] = (__bf16)fmaxf(c[j][v] * sc[m] + sh[m], 0.0f);
    }
    *(v8bf*)(fout + ((size_t)n * kHW + pix) * kD + m0 + 8 * hi) = o;  // 16B store
  }
}

// ---- DHT: segment-sum over pixels into (angle, rho) bins -------------------
// f[n][hw][c] channels-last: 16 lanes share a pixel (32B of 16 channels).
__global__ void dht_kernel(const __bf16* __restrict__ f, __bf16* __restrict__ acc) {
  __shared__ float hist[kR][16];
  const int bid = blockIdx.x;
  const int a  = bid % kA;
  const int n  = (bid / kA) % kN;
  const int c0 = (bid / (kA * kN)) * 16;
  const int tid = threadIdx.x;
  const int ch = tid & 15;

  for (int i = tid; i < kR * 16; i += blockDim.x) ((float*)hist)[i] = 0.0f;
  __syncthreads();

  const float irho = 182.0f / 179.0f;   // int(sqrt(H^2+W^2)+1)/(R-1)
  float sn, cs;
  __sincosf((float)a * (3.14159265358979323846f / 180.0f), &sn, &cs);
  const float tc = cs / irho, ts = sn / irho;

  const __bf16* fp = f + (size_t)n * kHW * kD + c0 + ch;
  for (int pix = tid >> 4; pix < kHW; pix += 16) {
    __builtin_prefetch(fp + (size_t)(pix + 16) * kD, 0, 0);  // global_prefetch_b8
    const int px = (pix & (kW - 1)) - kW / 2;
    const int py = (pix >> 7) - kH / 2;
    int r = (int)rintf(tc * (float)px + ts * (float)py) + kR / 2;
    r = min(max(r, 0), kR - 1);
    atomicAdd(&hist[r][ch], (float)fp[(size_t)pix * kD]);    // ds_add_f32
  }
  __syncthreads();

  for (int i = tid; i < kR * 16; i += blockDim.x) {
    const int r = i >> 4, ch2 = i & 15;
    acc[((size_t)(n * kA + a) * kR + r) * kD + c0 + ch2] = (__bf16)hist[r][ch2];
  }
}

// ---- conv3x3 implicit-GEMM, block = (n, a, 64-col group) x 8 cout waves ----
// Halo slab 3 rows x 66 cols x 128 cin staged in LDS (zero-filled OOB),
// padded col stride 136 (68 dwords) -> conflict-free ds_load_b128.
// Main loop: 9 taps x 4 k-chunks x 4 N-tiles = 144 WMMAs, branch-free.
template <bool OUTF32>
__global__ void __launch_bounds__(256)
conv3x3_wmma(const __bf16* __restrict__ in, const __bf16* __restrict__ wp,
             const float* __restrict__ sc, const float* __restrict__ sh,
             __bf16* __restrict__ outb, float* __restrict__ outf) {
  __shared__ __bf16 slab[3][66][136];              // 53.9 KB
  const int tid = threadIdx.x;
  const int g = blockIdx.x % 3;                    // 3 groups of 64 rho cols
  const int a = (blockIdx.x / 3) % kA;
  const int n = blockIdx.x / (3 * kA);
  const int colBase = g * 64 - 1;

  for (int idx = tid; idx < 3 * 66 * 16; idx += 256) {       // v8bf chunks
    const int k8  = (idx & 15) * 8;
    const int c16 = (idx >> 4) % 66;
    const int row = (idx >> 4) / 66;
    const int ar  = a + row - 1;
    const int col = colBase + c16;
    v8bf val = {};
    if ((unsigned)ar < (unsigned)kA && (unsigned)col < (unsigned)kR)
      val = *(const v8bf*)(in + ((size_t)(n * kA + ar) * kR + col) * kD + k8);
    *(v8bf*)&slab[row][c16][k8] = val;
  }
  __syncthreads();

  const int wave = tid >> 5, lane = tid & 31;
  const int m0 = wave << 4;
  const int hi = lane >> 4, colj = lane & 15;

  v8f c[4] = {{}, {}, {}, {}};
#pragma unroll
  for (int t = 0; t < 9; ++t) {
    const int ky = t / 3, kx = t % 3;
    const __bf16* wrow = wp + ((size_t)t * kD + m0 + colj) * kD + hi * 8;
#pragma unroll
    for (int kc = 0; kc < kD / 32; ++kc) {
      const int k0 = kc * 32;
      v16bf av = frag16(wrow + k0);
#pragma unroll
      for (int j = 0; j < 4; ++j) {
        v16bf bv = frag16(&slab[ky][colj + j * 16 + kx][k0 + hi * 8]);
        c[j] = __builtin_amdgcn_wmma_f32_16x16x32_bf16(false, av, false, bv,
                                                       (short)0, c[j], false, false);
      }
    }
  }

#pragma unroll
  for (int j = 0; j < 4; ++j) {
    const int colo = g * 64 + j * 16 + colj;
    if (colo >= kR) continue;
    if (OUTF32) {
#pragma unroll
      for (int v = 0; v < 8; ++v) {     // d_out is NCHW f32: scattered b32
        const int m = m0 + v + 8 * hi;
        outf[(((size_t)n * kD + m) * kA + a) * kR + colo] =
            fmaxf(c[j][v] * sc[m] + sh[m], 0.0f);
      }
    } else {
      v8bf o;
#pragma unroll
      for (int v = 0; v < 8; ++v) {
        const int m = m0 + v + 8 * hi;
        o[v] = (__bf16)fmaxf(c[j][v] * sc[m] + sh[m], 0.0f);
      }
      *(v8bf*)(outb + ((size_t)(n * kA + a) * kR + colo) * kD + m0 + 8 * hi) = o;
    }
  }
}

// ---------------------------------------------------------------------------
extern "C" void kernel_launch(void* const* d_in, const int* in_sizes, int n_in,
                              void* d_out, int out_size, void* d_ws, size_t ws_size,
                              hipStream_t stream) {
  (void)in_sizes; (void)n_in; (void)out_size; (void)ws_size;
  const float* x   = (const float*)d_in[0];
  const float* w1  = (const float*)d_in[1];
  const float* b1  = (const float*)d_in[2];
  const float* g1  = (const float*)d_in[3];
  const float* be1 = (const float*)d_in[4];
  const float* m1  = (const float*)d_in[5];
  const float* v1  = (const float*)d_in[6];
  const float* w2  = (const float*)d_in[7];
  const float* b2  = (const float*)d_in[8];
  const float* g2  = (const float*)d_in[9];
  const float* be2 = (const float*)d_in[10];
  const float* m2  = (const float*)d_in[11];
  const float* v2  = (const float*)d_in[12];
  const float* w3  = (const float*)d_in[13];
  const float* b3  = (const float*)d_in[14];
  const float* g3  = (const float*)d_in[15];
  const float* be3 = (const float*)d_in[16];
  const float* m3  = (const float*)d_in[17];
  const float* v3  = (const float*)d_in[18];

  char* ws = (char*)d_ws;
  size_t off = 0;
  auto carve = [&](size_t bytes) -> void* {
    void* p = ws + off;
    off += (bytes + 255) & ~(size_t)255;
    return p;
  };
  __bf16* xtb  = (__bf16*)carve((size_t)kN * kHW * kCi * 2);  // 64 MB  [n][hw][ci]
  __bf16* fbuf = (__bf16*)carve((size_t)kN * kHW * kD * 2);   // 33.5MB [n][hw][c]
  __bf16* accb = (__bf16*)carve((size_t)kN * kAR * kD * 2);   // 66.4MB [n][a][r][c]
  __bf16* y1b  = (__bf16*)carve((size_t)kN * kAR * kD * 2);   // 66.4MB [n][a][r][c]
  __bf16* w1b  = (__bf16*)carve((size_t)kD * kCi * 2);
  __bf16* w2p  = (__bf16*)carve((size_t)9 * kD * kD * 2);
  __bf16* w3p  = (__bf16*)carve((size_t)9 * kD * kD * 2);
  float*  sc1  = (float*)carve(kD * 4);
  float*  sh1  = (float*)carve(kD * 4);
  float*  sc2  = (float*)carve(kD * 4);
  float*  sh2  = (float*)carve(kD * 4);
  float*  sc3  = (float*)carve(kD * 4);
  float*  sh3  = (float*)carve(kD * 4);

  pack_f32_to_bf16<<<(kD * kCi + 255) / 256, 256, 0, stream>>>(w1, w1b, kD * kCi);
  pack_w3x3<<<(kD * kD * 9 + 255) / 256, 256, 0, stream>>>(w2, w2p);
  pack_w3x3<<<(kD * kD * 9 + 255) / 256, 256, 0, stream>>>(w3, w3p);
  bn_precompute<<<1, kD, 0, stream>>>(b1, g1, be1, m1, v1, sc1, sh1);
  bn_precompute<<<1, kD, 0, stream>>>(b2, g2, be2, m2, v2, sc2, sh2);
  bn_precompute<<<1, kD, 0, stream>>>(b3, g3, be3, m3, v3, sc3, sh3);

  transpose_x<<<kN * (kCi / 32) * (kHW / 32), 256, 0, stream>>>(x, xtb);

  // 8 batches * 256 pixel groups; 8 waves = 8 cout tiles per block
  conv1_wmma<<<kN * 256, 256, 0, stream>>>(xtb, w1b, sc1, sh1, fbuf);

  // 180 angles * 8 batches * 8 channel tiles
  dht_kernel<<<kA * kN * (kD / 16), 256, 0, stream>>>(fbuf, accb);

  // 3 col groups * 180 angles * 8 batches; 8 waves = 8 cout tiles per block
  conv3x3_wmma<false><<<3 * kA * kN, 256, 0, stream>>>(accb, w2p, sc2, sh2,
                                                       y1b, nullptr);
  conv3x3_wmma<true><<<3 * kA * kN, 256, 0, stream>>>(y1b, w3p, sc3, sh3,
                                                      nullptr, (float*)d_out);
}
